// GraphTransformerNet_55104430408159
// MI455X (gfx1250) — matrix operand order repaired and checked by reading, not compiled, not run on hardware
//
#include <hip/hip_runtime.h>
#include <hip/hip_bf16.h>

// ---------------------------------------------------------------------------
// Problem constants (match the reference)
// ---------------------------------------------------------------------------
constexpr int C_B   = 128;              // graphs
constexpr int C_NN  = 128;              // nodes per graph
constexpr int C_NF  = 10;               // node features
constexpr int C_EF  = 2;                // edge features
constexpr int C_D   = 128;              // d_model
constexpr int C_L   = 4;                // layers
constexpr int C_H   = 8;                // heads
constexpr int C_DK  = 16;               // head dim
constexpr int C_DFF = 512;              // policy hidden
constexpr int C_N   = C_B * C_NN;       // 16384 nodes total
constexpr int C_M   = C_N * 8;          // 131072 edges total

typedef __attribute__((ext_vector_type(16))) _Float16 v16h;
typedef __attribute__((ext_vector_type(4)))  _Float16 v4h;
typedef __attribute__((ext_vector_type(8)))  float    v8f;

static inline int cdiv(int a, int b) { return (a + b - 1) / b; }

// ---------------------------------------------------------------------------
// f32 -> f16 weight conversion (one-time per launch; weights total ~1M elems)
// ---------------------------------------------------------------------------
__global__ __launch_bounds__(256)
void f32_to_f16(const float* __restrict__ s, _Float16* __restrict__ d, int n)
{
    const int idx = blockIdx.x * blockDim.x + threadIdx.x;
    if (idx < n) d[idx] = (_Float16)s[idx];
}

// ---------------------------------------------------------------------------
// WMMA GEMM:  C[Mr x Nc] = A[Mr x K](f32) @ B[K x Nc](f16)  (+bias)(+resid)(relu)
// Block = 256 threads (8 waves) computing a 128x64 tile:
//   wave w -> rows [bm*128 + w*16, +16), cols [bn*64, +64)  == 4 WMMA tiles.
// B K-chunk (32x64) is staged in LDS pre-swizzled to the CDNA5 B-fragment
// layout (ISA 7.12.2) and double-buffered; each lane reads its fragment as a
// single 32-byte LDS vector. A fragment is 4x global_load_b128 + cvt per step;
// next step's A is prefetched (global_prefetch_b8). All four B fragments are
// loaded before the four WMMAs so they issue back-to-back on independent accs.
// Requires: Mr % 128 == 0, Nc % 64 == 0, K % 32 == 0 (true for every call).
// ---------------------------------------------------------------------------
template<bool BIAS, bool RESID, bool RELU>
__global__ __launch_bounds__(256)
void gemm_wmma(const float* __restrict__ A, const _Float16* __restrict__ B,
               const float* __restrict__ bias, const float* __restrict__ resid,
               float* __restrict__ C, int Mr, int Nc, int K)
{
    const int nBlkN = Nc >> 6;
    const int bm    = blockIdx.x / nBlkN;
    const int bn    = blockIdx.x - bm * nBlkN;
    const int row0  = bm * 128;
    const int col0  = bn * 64;

    const int tid  = threadIdx.x;
    const int lane = tid & 31;
    const int wave = tid >> 5;
    const int hi   = lane >> 4;              // 0: lanes 0-15, 1: lanes 16-31
    const int l15  = lane & 15;

    // [buf][coltile][fragment-lane][16 halves] ; one lane row = 32B
    __shared__ __align__(32) _Float16 Bs[2][4][32][16];

    const int    mrow = row0 + wave * 16 + l15;   // A row held by this lane
    const float* Arow = A + (size_t)mrow * K;

    v8f acc[4] = {{}, {}, {}, {}};

    for (int kk = 0, it = 0; kk < K; kk += 32, ++it) {
        // ---- stage B[kk..kk+32)[col0..col0+64) into LDS (swizzled) --------
        _Float16 (*bs)[32][16] = Bs[it & 1];
#pragma unroll
        for (int rep = 0; rep < 2; ++rep) {
            const int quad = tid + rep * 256;      // 512 quads of 4 halves
            const int k    = quad >> 4;            // 0..31
            const int n4   = (quad & 15) * 4;      // 0..60
            const v4h q = *(const v4h*)(B + (size_t)(kk + k) * Nc + col0 + n4);
#pragma unroll
            for (int j = 0; j < 4; ++j) {
                const int n = n4 + j;
                bs[n >> 4][(n & 15) + (((k >> 3) & 1) << 4)][(k & 7) + ((k >> 4) << 3)] = q[j];
            }
        }
        __syncthreads();

        // ---- A fragment: lanes 0-15 K {0..7,16..23}, lanes 16-31 +8 -------
        const int ak = kk + hi * 8;
        const float4 a0 = *(const float4*)(Arow + ak);
        const float4 a1 = *(const float4*)(Arow + ak + 4);
        const float4 a2 = *(const float4*)(Arow + ak + 16);
        const float4 a3 = *(const float4*)(Arow + ak + 20);
        if (kk + 32 < K) {
            __builtin_prefetch(Arow + ak + 32, 0, 3);   // next A chunk
            __builtin_prefetch(Arow + ak + 48, 0, 3);
        }
        v16h a;
        a[0] = (_Float16)a0.x; a[1] = (_Float16)a0.y; a[2] = (_Float16)a0.z; a[3] = (_Float16)a0.w;
        a[4] = (_Float16)a1.x; a[5] = (_Float16)a1.y; a[6] = (_Float16)a1.z; a[7] = (_Float16)a1.w;
        a[8]  = (_Float16)a2.x; a[9]  = (_Float16)a2.y; a[10] = (_Float16)a2.z; a[11] = (_Float16)a2.w;
        a[12] = (_Float16)a3.x; a[13] = (_Float16)a3.y; a[14] = (_Float16)a3.z; a[15] = (_Float16)a3.w;

        // ---- load all four B fragments, then 4 back-to-back WMMAs ---------
        const v16h b0 = *(const v16h*)(&bs[0][lane][0]);
        const v16h b1 = *(const v16h*)(&bs[1][lane][0]);
        const v16h b2 = *(const v16h*)(&bs[2][lane][0]);
        const v16h b3 = *(const v16h*)(&bs[3][lane][0]);
        acc[0] = __builtin_amdgcn_wmma_f32_16x16x32_f16(false, a, false, b0, (short)0, acc[0], false, false);
        acc[1] = __builtin_amdgcn_wmma_f32_16x16x32_f16(false, a, false, b1, (short)0, acc[1], false, false);
        acc[2] = __builtin_amdgcn_wmma_f32_16x16x32_f16(false, a, false, b2, (short)0, acc[2], false, false);
        acc[3] = __builtin_amdgcn_wmma_f32_16x16x32_f16(false, a, false, b3, (short)0, acc[3], false, false);
    }

    // ---- epilogue: f32 C layout (lanes 0-15 rows 0..7, 16-31 rows 8..15) --
#pragma unroll
    for (int tc = 0; tc < 4; ++tc) {
        const int ncol = col0 + tc * 16 + l15;
        const float bv = BIAS ? bias[ncol] : 0.0f;
#pragma unroll
        for (int r = 0; r < 8; ++r) {
            const int m = row0 + wave * 16 + hi * 8 + r;
            float v = acc[tc][r] + bv;
            if (RESID) v += resid[(size_t)m * Nc + ncol];
            if (RELU)  v = fmaxf(v, 0.0f);
            C[(size_t)m * Nc + ncol] = v;
        }
    }
}

// ---------------------------------------------------------------------------
// Tiny-K embedding GEMM (K = 10 or 2): plain VALU, one thread per output elem
// ---------------------------------------------------------------------------
__global__ __launch_bounds__(256)
void embed_kernel(const float* __restrict__ X, const float* __restrict__ W,
                  const float* __restrict__ bvec, float* __restrict__ Y,
                  int R, int Kin, int Dout)
{
    const int idx = blockIdx.x * blockDim.x + threadIdx.x;
    if (idx >= R * Dout) return;
    const int r = idx / Dout, d = idx - r * Dout;
    float s = bvec[d];
    for (int k = 0; k < Kin; ++k) s += X[(size_t)r * Kin + k] * W[(size_t)k * Dout + d];
    Y[idx] = s;
}

// ---------------------------------------------------------------------------
// Edge attention: one thread per (edge, head).
// ---------------------------------------------------------------------------
__global__ __launch_bounds__(256)
void edge_attn(const float* __restrict__ Q, const float* __restrict__ Km,
               const float* __restrict__ V, const float* __restrict__ E,
               const int* __restrict__ src, const int* __restrict__ dst,
               float* __restrict__ e_att, float* __restrict__ wV,
               float* __restrict__ z)
{
    const int idx = blockIdx.x * blockDim.x + threadIdx.x;
    if (idx >= C_M * C_H) return;
    const int m  = idx >> 3;
    const int hh = idx & 7;
    const int s  = src[m];
    const int d  = dst[m];
    const int off = hh * C_DK;
    const float inv = 0.25f;                    // 1/sqrt(16)

    float ssum = 0.0f;
#pragma unroll
    for (int k = 0; k < C_DK; ++k) {
        const float sc = Km[(size_t)s * C_D + off + k] *
                         Q [(size_t)d * C_D + off + k] * inv *
                         E [(size_t)m * C_D + off + k];
        e_att[(size_t)m * C_D + off + k] = sc;
        ssum += sc;
    }
    const float w = __expf(fminf(fmaxf(ssum, -5.0f), 5.0f));
#pragma unroll
    for (int k = 0; k < C_DK; ++k)
        atomicAdd(&wV[(size_t)d * C_D + off + k],
                  w * V[(size_t)s * C_D + off + k]);
    atomicAdd(&z[(size_t)d * C_H + hh], w);
}

// h_att = wV / (z + 1e-6)  (in place over wV is safe: 1:1 element map)
__global__ __launch_bounds__(256)
void attn_div(const float* __restrict__ wV, const float* __restrict__ z,
              float* __restrict__ h_att)
{
    const int idx = blockIdx.x * blockDim.x + threadIdx.x;
    if (idx >= C_N * C_D) return;
    const int n  = idx >> 7;
    const int dd = idx & 127;
    h_att[idx] = wV[idx] / (z[(size_t)n * C_H + (dd >> 4)] + 1e-6f);
}

// ---------------------------------------------------------------------------
// BatchNorm (training-mode batch stats over dim 0)
// ---------------------------------------------------------------------------
__global__ __launch_bounds__(256)
void bn_stats(const float* __restrict__ X, int R, int Cc,
              float* __restrict__ mu, float* __restrict__ rstd)
{
    const int c = blockIdx.x;
    float s = 0.0f, s2 = 0.0f;
    for (int r = threadIdx.x; r < R; r += blockDim.x) {
        const float v = X[(size_t)r * Cc + c];
        s += v; s2 += v * v;
    }
    __shared__ float sh[256], sh2[256];
    sh[threadIdx.x] = s; sh2[threadIdx.x] = s2;
    __syncthreads();
    for (int st = 128; st > 0; st >>= 1) {
        if (threadIdx.x < st) {
            sh[threadIdx.x]  += sh[threadIdx.x + st];
            sh2[threadIdx.x] += sh2[threadIdx.x + st];
        }
        __syncthreads();
    }
    if (threadIdx.x == 0) {
        const float mean = sh[0] / (float)R;
        const float var  = sh2[0] / (float)R - mean * mean;
        mu[c]   = mean;
        rstd[c] = rsqrtf(var + 1e-5f);
    }
}

__global__ __launch_bounds__(256)
void bn_apply(const float* __restrict__ X, const float* __restrict__ mu,
              const float* __restrict__ rstd, const float* __restrict__ g,
              const float* __restrict__ b, float* __restrict__ Y, int R, int Cc)
{
    const int idx = blockIdx.x * blockDim.x + threadIdx.x;
    if (idx >= R * Cc) return;
    const int c = idx % Cc;
    Y[idx] = g[c] * (X[idx] - mu[c]) * rstd[c] + b[c];
}

// pairs[n] = concat(h[veh_global[n/NN]], h[n])   (N x 2D)
__global__ __launch_bounds__(256)
void pairs_gather(const float* __restrict__ hb, const int* __restrict__ veh,
                  float* __restrict__ pairs)
{
    const int idx = blockIdx.x * blockDim.x + threadIdx.x;
    if (idx >= C_N * 2 * C_D) return;
    const int n  = idx >> 8;
    const int dd = idx & 255;
    const int bg = n >> 7;
    const int vg = bg * C_NN + veh[bg];
    pairs[idx] = (dd < C_D) ? hb[(size_t)vg * C_D + dd]
                            : hb[(size_t)n  * C_D + (dd - C_D)];
}

// policy[n] = dot(m1[n,:512], Wm2) + bm2 ; one wave per row
__global__ __launch_bounds__(256)
void policy_out(const float* __restrict__ m1, const float* __restrict__ Wm2,
                const float* __restrict__ bm2, float* __restrict__ outp)
{
    const int wid  = (blockIdx.x * blockDim.x + threadIdx.x) >> 5;
    const int lane = threadIdx.x & 31;
    if (wid >= C_N) return;
    float s = 0.0f;
    for (int k = lane; k < C_DFF; k += 32)
        s += m1[(size_t)wid * C_DFF + k] * Wm2[k];
#pragma unroll
    for (int off = 16; off > 0; off >>= 1)
        s += __shfl_down(s, off, 32);
    if (lane == 0) outp[wid] = s + bm2[0];
}

// ---------------------------------------------------------------------------
// Host-side orchestration
// ---------------------------------------------------------------------------
extern "C" void kernel_launch(void* const* d_in, const int* in_sizes, int n_in,
                              void* d_out, int out_size, void* d_ws, size_t ws_size,
                              hipStream_t stream)
{
    const float* h_in   = (const float*)d_in[0];
    const float* e_in   = (const float*)d_in[1];
    const int*   src    = (const int*)  d_in[2];
    const int*   dst    = (const int*)  d_in[3];
    const int*   veh    = (const int*)  d_in[4];
    const float* WembH  = (const float*)d_in[6];
    const float* bembH  = (const float*)d_in[7];
    const float* WembE  = (const float*)d_in[8];
    const float* bembE  = (const float*)d_in[9];
    const float* Wq     = (const float*)d_in[10];
    const float* Wk     = (const float*)d_in[11];
    const float* Wv     = (const float*)d_in[12];
    const float* We     = (const float*)d_in[13];
    const float* WoH    = (const float*)d_in[14];
    const float* WoE    = (const float*)d_in[15];
    const float* boH    = (const float*)d_in[16];
    const float* boE    = (const float*)d_in[17];
    const float* g1h    = (const float*)d_in[18];
    const float* g1e    = (const float*)d_in[19];
    const float* g2h    = (const float*)d_in[20];
    const float* g2e    = (const float*)d_in[21];
    const float* b1h    = (const float*)d_in[22];
    const float* b1e    = (const float*)d_in[23];
    const float* b2h    = (const float*)d_in[24];
    const float* b2e    = (const float*)d_in[25];
    const float* Wf1h   = (const float*)d_in[26];
    const float* bf1h   = (const float*)d_in[27];
    const float* Wf2h   = (const float*)d_in[28];
    const float* bf2h   = (const float*)d_in[29];
    const float* Wf1e   = (const float*)d_in[30];
    const float* bf1e   = (const float*)d_in[31];
    const float* Wf2e   = (const float*)d_in[32];
    const float* bf2e   = (const float*)d_in[33];
    const float* Wm1    = (const float*)d_in[34];
    const float* bm1    = (const float*)d_in[35];
    const float* Wm2    = (const float*)d_in[36];
    const float* bm2    = (const float*)d_in[37];

    const size_t ND  = (size_t)C_N * C_D;   // 2,097,152
    const size_t MD  = (size_t)C_M * C_D;   // 16,777,216
    const size_t DD  = (size_t)C_L * C_D * C_D;       // stacked DxD weights
    const size_t DD2 = (size_t)C_L * C_D * 2 * C_D;   // stacked Dx2D weights

    float* ws = (float*)d_ws;
    size_t off = 0;
    auto wsalloc = [&](size_t n) { float* p = ws + off; off += n; return p; };

    float* hbuf = wsalloc(ND);        // node state
    float* ebuf = wsalloc(MD);        // edge state
    float* qb   = wsalloc(ND);
    float* kb   = wsalloc(ND);
    float* vb   = wsalloc(ND);
    float* Ebuf = wsalloc(MD);        // E projection; later e1; later m1
    float* te   = wsalloc(MD);        // edge tmp (pre-BN)
    float* ef1  = wsalloc(2 * MD);    // edge FFN hidden; e_att aliases base
    float* eatt = ef1;
    float* wvb  = wsalloc(ND);        // scatter accum; becomes h_att in place
    float* zbuf = wsalloc((size_t)C_N * C_H);
    float* th   = wsalloc(ND);        // node tmp (pre-BN)
    float* h1   = wsalloc(ND);
    float* hf1  = wsalloc(2 * ND);    // node FFN hidden; pairs aliases it
    float* mu_h = wsalloc(C_D);
    float* rs_h = wsalloc(C_D);
    float* mu_e = wsalloc(C_D);
    float* rs_e = wsalloc(C_D);

    // f16 weight arena (bump-allocated in halves)
    _Float16* harena = (_Float16*)wsalloc((6 * DD + 4 * DD2 + 2 * C_D * C_DFF) / 2 + 64);
    size_t hoff = 0;
    auto cvt = [&](const float* srcw, size_t n) {
        _Float16* p = harena + hoff; hoff += n;
        f32_to_f16<<<cdiv((int)n, 256), 256, 0, stream>>>(srcw, p, (int)n);
        return p;
    };
    _Float16* hWq   = cvt(Wq,   DD);
    _Float16* hWk   = cvt(Wk,   DD);
    _Float16* hWv   = cvt(Wv,   DD);
    _Float16* hWe   = cvt(We,   DD);
    _Float16* hWoH  = cvt(WoH,  DD);
    _Float16* hWoE  = cvt(WoE,  DD);
    _Float16* hWf1h = cvt(Wf1h, DD2);
    _Float16* hWf2h = cvt(Wf2h, DD2);
    _Float16* hWf1e = cvt(Wf1e, DD2);
    _Float16* hWf2e = cvt(Wf2e, DD2);
    _Float16* hWm1  = cvt(Wm1,  (size_t)2 * C_D * C_DFF);

    auto gemm = [&](const float* A, const _Float16* Bh, const float* bias,
                    const float* resid, float* C, int Mr, int Nc, int K, int relu) {
        const dim3 grid((Mr / 128) * (Nc / 64));
        if (bias && resid)
            gemm_wmma<true, true, false><<<grid, 256, 0, stream>>>(A, Bh, bias, resid, C, Mr, Nc, K);
        else if (bias && relu)
            gemm_wmma<true, false, true><<<grid, 256, 0, stream>>>(A, Bh, bias, nullptr, C, Mr, Nc, K);
        else
            gemm_wmma<false, false, false><<<grid, 256, 0, stream>>>(A, Bh, nullptr, nullptr, C, Mr, Nc, K);
    };

    // ---- embeddings -------------------------------------------------------
    embed_kernel<<<cdiv(C_N * C_D, 256), 256, 0, stream>>>(h_in, WembH, bembH, hbuf, C_N, C_NF, C_D);
    embed_kernel<<<cdiv(C_M * C_D, 256), 256, 0, stream>>>(e_in, WembE, bembE, ebuf, C_M, C_EF, C_D);

    // ---- transformer layers ----------------------------------------------
    for (int l = 0; l < C_L; ++l) {
        const size_t oDD = (size_t)l * C_D * C_D;
        const size_t oD  = (size_t)l * C_D;
        const size_t oD2 = (size_t)l * C_D * 2 * C_D;
        const size_t o2D = (size_t)l * 2 * C_D;

        // projections
        gemm(hbuf, hWq + oDD, nullptr, nullptr, qb,   C_N, C_D, C_D, 0);
        gemm(hbuf, hWk + oDD, nullptr, nullptr, kb,   C_N, C_D, C_D, 0);
        gemm(hbuf, hWv + oDD, nullptr, nullptr, vb,   C_N, C_D, C_D, 0);
        gemm(ebuf, hWe + oDD, nullptr, nullptr, Ebuf, C_M, C_D, C_D, 0);

        // attention (scatter accumulators zeroed each layer)
        hipMemsetAsync(wvb,  0, ND * sizeof(float), stream);
        hipMemsetAsync(zbuf, 0, (size_t)C_N * C_H * sizeof(float), stream);
        edge_attn<<<cdiv(C_M * C_H, 256), 256, 0, stream>>>(qb, kb, vb, Ebuf, src, dst,
                                                            eatt, wvb, zbuf);
        attn_div<<<cdiv(C_N * C_D, 256), 256, 0, stream>>>(wvb, zbuf, wvb);

        // O projections + residual, then BatchNorm
        gemm(wvb,  hWoH + oDD, boH + oD, hbuf, th, C_N, C_D, C_D, 0);
        gemm(eatt, hWoE + oDD, boE + oD, ebuf, te, C_M, C_D, C_D, 0);
        bn_stats<<<C_D, 256, 0, stream>>>(th, C_N, C_D, mu_h, rs_h);
        bn_apply<<<cdiv(C_N * C_D, 256), 256, 0, stream>>>(th, mu_h, rs_h,
                                                           g1h + oD, b1h + oD, h1, C_N, C_D);
        bn_stats<<<C_D, 256, 0, stream>>>(te, C_M, C_D, mu_e, rs_e);
        bn_apply<<<cdiv(C_M * C_D, 256), 256, 0, stream>>>(te, mu_e, rs_e,
                                                           g1e + oD, b1e + oD, Ebuf, C_M, C_D);
        float* e1 = Ebuf;

        // FFN (node): relu(h1@Wf1+b) @ Wf2 + b + h1, then BN -> hbuf
        gemm(h1,  hWf1h + oD2, bf1h + o2D, nullptr, hf1, C_N, 2 * C_D, C_D, 1);
        gemm(hf1, hWf2h + oD2, bf2h + oD,  h1,      th,  C_N, C_D, 2 * C_D, 0);
        bn_stats<<<C_D, 256, 0, stream>>>(th, C_N, C_D, mu_h, rs_h);
        bn_apply<<<cdiv(C_N * C_D, 256), 256, 0, stream>>>(th, mu_h, rs_h,
                                                           g2h + oD, b2h + oD, hbuf, C_N, C_D);

        // FFN (edge): relu(e1@Wf1+b) @ Wf2 + b + e1, then BN -> ebuf
        gemm(e1,  hWf1e + oD2, bf1e + o2D, nullptr, ef1, C_M, 2 * C_D, C_D, 1);
        gemm(ef1, hWf2e + oD2, bf2e + oD,  e1,      te,  C_M, C_D, 2 * C_D, 0);
        bn_stats<<<C_D, 256, 0, stream>>>(te, C_M, C_D, mu_e, rs_e);
        bn_apply<<<cdiv(C_M * C_D, 256), 256, 0, stream>>>(te, mu_e, rs_e,
                                                           g2e + oD, b2e + oD, ebuf, C_M, C_D);
    }

    // ---- policy head ------------------------------------------------------
    float* pairs = hf1;               // N x 2D
    float* m1    = Ebuf;              // N x DFF (8.4M floats fits in MD slot)
    pairs_gather<<<cdiv(C_N * 2 * C_D, 256), 256, 0, stream>>>(hbuf, veh, pairs);
    gemm(pairs, hWm1, bm1, nullptr, m1, C_N, C_DFF, 2 * C_D, 1);
    policy_out<<<cdiv(C_N * 32, 256), 256, 0, stream>>>(m1, Wm2, bm2, (float*)d_out);
}
